// EdgeAttentionEncoder_51728586113560
// MI455X (gfx1250) — matrix operand order repaired and checked by reading, not compiled, not run on hardware
//
#include <hip/hip_runtime.h>

// ---------------------------------------------------------------------------
// EdgeAttentionEncoder (triangle attention), MI455X / gfx1250, wave32 + WMMA
// B=2, N=64, D=256, H=8, DK=32
// f16-staged operands: pack once, GEMM loops are pure b128-load + v_wmma.
// ---------------------------------------------------------------------------

typedef __attribute__((ext_vector_type(16))) _Float16 v16h;
typedef __attribute__((ext_vector_type(8)))  _Float16 v8h;
typedef __attribute__((ext_vector_type(4)))  _Float16 v4h;
typedef __attribute__((ext_vector_type(8)))  float    v8f;

#define B_    2
#define N_    64
#define D_    256
#define H_    8
#define DK_   32
#define ROWS_ (B_ * N_ * N_)          // 8192 flattened (b,x,y) rows
#define NN_   (N_ * N_)               // 4096
#define DD_   (D_ * D_)               // 65536 (one weight matrix)

// ---- WMMA fragment layout (wave32) ----------------------------------------
// A (16x32 f16): lane L holds row M=L%16; half=L/16:
//   half 0: elems 0..7 = K 0..7,  elems 8..15 = K 16..23
//   half 1: elems 0..7 = K 8..15, elems 8..15 = K 24..31
// B (32x16 f16): lane L holds column N=L%16;
//   half 0: elems 0..7 = K 0..7, elems 8..15 = K 16..23 ; half 1: K 8..15, K 24..31
// Both load as two 16-byte (8-half) chunks -> 2x global_load_b128 each.

__device__ __forceinline__ v16h join16(v8h lo, v8h hi) {
  return __builtin_shufflevector(lo, hi, 0, 1, 2, 3, 4, 5, 6, 7,
                                          8, 9, 10, 11, 12, 13, 14, 15);
}

__device__ __forceinline__ v16h load_a_frag(const _Float16* __restrict__ row,
                                            int k0, int half) {
  const v8h lo = *(const v8h*)(row + k0 + half * 8);
  const v8h hi = *(const v8h*)(row + k0 + 16 + half * 8);
  return join16(lo, hi);
}

__device__ __forceinline__ v16h load_b_frag(const _Float16* __restrict__ row,
                                            int k0, int half) {
  const v8h lo = *(const v8h*)(row + k0 + half * 8);
  const v8h hi = *(const v8h*)(row + k0 + 16 + half * 8);
  return join16(lo, hi);
}

__device__ __forceinline__ v8f wmma_f16(v16h a, v16h b, v8f c) {
  return __builtin_amdgcn_wmma_f32_16x16x32_f16(false, a, false, b, (short)0, c,
                                                false, false);
}

// ---------------------------------------------------------------------------
// Kernel 0: pack fp32 -> f16 (streaming, L2-resident). 4 floats per thread.
// ---------------------------------------------------------------------------
__global__ void pack_f16_kernel(const float* __restrict__ src,
                                _Float16* __restrict__ dst, int n4) {
  const int i = blockIdx.x * blockDim.x + threadIdx.x;
  if (i < n4) {
    const float4 f = ((const float4*)src)[i];
    v4h h;
    h[0] = (_Float16)f.x; h[1] = (_Float16)f.y;
    h[2] = (_Float16)f.z; h[3] = (_Float16)f.w;
    ((v4h*)dst)[i] = h;
  }
}

// ---------------------------------------------------------------------------
// Kernel 1: four projections  y = state @ W^T + b   ([8192,256] x [256,256])
// One wave per 16x16 tile; shared A fragment feeds 4 WMMAs per k-step.
// lk/rk stored f16 (feed score WMMA), lv/rv stored f32 (feed triple product).
// ---------------------------------------------------------------------------
__global__ void proj4_kernel(const _Float16* __restrict__ state16,
                             const _Float16* __restrict__ Wlk16,
                             const _Float16* __restrict__ Wrk16,
                             const _Float16* __restrict__ Wlv16,
                             const _Float16* __restrict__ Wrv16,
                             const float* __restrict__ blk,
                             const float* __restrict__ brk,
                             const float* __restrict__ blv,
                             const float* __restrict__ brv,
                             _Float16* __restrict__ lk16,
                             _Float16* __restrict__ rk16,
                             float* __restrict__ lv32,
                             float* __restrict__ rv32) {
  const int wave = (blockIdx.x * blockDim.x + threadIdx.x) >> 5;
  const int lane = threadIdx.x & 31;
  const int half = lane >> 4;
  const int l16  = lane & 15;

  const int tile_n = wave & 15;          // 16 column tiles (D/16)
  const int tile_m = wave >> 4;          // 512 row tiles (ROWS/16)

  const int arow = tile_m * 16 + l16;    // A row (lane's matrix row)
  const int ecol = tile_n * 16 + l16;    // output feature = W row (lane's col)

  const _Float16* aRow = state16 + (size_t)arow * D_;
  const _Float16* w0   = Wlk16 + (size_t)ecol * D_;
  const _Float16* w1   = Wrk16 + (size_t)ecol * D_;
  const _Float16* w2   = Wlv16 + (size_t)ecol * D_;
  const _Float16* w3   = Wrv16 + (size_t)ecol * D_;

  v8f c0 = {}, c1 = {}, c2 = {}, c3 = {};
#pragma unroll
  for (int kk = 0; kk < D_ / 32; ++kk) {
    const int k0 = kk * 32;
    const v16h a   = load_a_frag(aRow, k0, half);
    const v16h bf0 = load_b_frag(w0, k0, half);
    const v16h bf1 = load_b_frag(w1, k0, half);
    const v16h bf2 = load_b_frag(w2, k0, half);
    const v16h bf3 = load_b_frag(w3, k0, half);
    c0 = wmma_f16(a, bf0, c0);
    c1 = wmma_f16(a, bf1, c1);
    c2 = wmma_f16(a, bf2, c2);
    c3 = wmma_f16(a, bf3, c3);
  }

  const float bias0 = blk[ecol];
  const float bias1 = brk[ecol];
  const float bias2 = blv[ecol];
  const float bias3 = brv[ecol];

  // C/D layout: VGPR v, lanes 0-15 -> M=v ; lanes 16-31 -> M=v+8 ; col = l16
#pragma unroll
  for (int v = 0; v < 8; ++v) {
    const int orow = tile_m * 16 + half * 8 + v;
    const size_t idx = (size_t)orow * D_ + ecol;
    lk16[idx] = (_Float16)(c0[v] + bias0);
    rk16[idx] = (_Float16)(c1[v] + bias1);
    lv32[idx] = c2[v] + bias2;
    rv32[idx] = c3[v] + bias3;
  }
}

// ---------------------------------------------------------------------------
// Kernel 2: scores[b,h,a][x,y] = (1/sqrt(DK)) * sum_d lk[b,x,a,h,d]*rk[b,a,y,h,d]
// K = DK = 32 -> exactly one WMMA per 16x16 tile. Masked, stored [b,h,a,x,y].
// ---------------------------------------------------------------------------
__global__ void scores_kernel(const _Float16* __restrict__ lk16,
                              const _Float16* __restrict__ rk16,
                              const unsigned char* __restrict__ mask,
                              float* __restrict__ att) {
  const int wave = (blockIdx.x * blockDim.x + threadIdx.x) >> 5;
  const int lane = threadIdx.x & 31;
  const int half = lane >> 4;
  const int l16  = lane & 15;

  const int tile = wave & 15;            // 4x4 tiles of the 64x64 score matrix
  const int slab = wave >> 4;            // (b,h,a)
  const int a = slab & (N_ - 1);
  const int h = (slab >> 6) & (H_ - 1);
  const int b = slab >> 9;

  const int x0 = (tile >> 2) * 16;
  const int y0 = (tile & 3) * 16;

  const _Float16* aRow =
      lk16 + ((size_t)((b * N_ + (x0 + l16)) * N_ + a)) * D_ + h * DK_;
  const _Float16* bRow =
      rk16 + ((size_t)((b * N_ + a) * N_ + (y0 + l16))) * D_ + h * DK_;

  const v16h af = load_a_frag(aRow, 0, half);
  const v16h bf = load_b_frag(bRow, 0, half);
  v8f c = {};
  c = wmma_f16(af, bf, c);

  const float scale = 0.17677669529663687f;  // 1/sqrt(32)
#pragma unroll
  for (int v = 0; v < 8; ++v) {
    const int xe = x0 + half * 8 + v;
    const int ye = y0 + l16;
    float s = c[v] * scale;
    if (mask[((size_t)(b * N_ + xe) * N_ + a) * N_ + ye]) s = -1000.0f;
    att[((size_t)((b * H_ + h) * N_ + a)) * NN_ + xe * N_ + ye] = s;
  }
}

// ---------------------------------------------------------------------------
// Kernel 3: softmax over intermediate node a (stride NN in [b,h,a,x,y]).
// One thread per (b,h,x,y); 64 logits cached in registers; coalesced planes.
// ---------------------------------------------------------------------------
__global__ void softmax_kernel(float* __restrict__ att) {
  const int t  = blockIdx.x * blockDim.x + threadIdx.x;   // 65536 threads
  const int y  = t & (N_ - 1);
  const int x  = (t >> 6) & (N_ - 1);
  const int bh = t >> 12;                                 // 0..15
  float* base = att + (size_t)bh * N_ * NN_ + x * N_ + y;

  float v[N_];
  float m = -1e30f;
#pragma unroll
  for (int a = 0; a < N_; ++a) { v[a] = base[a * NN_]; m = fmaxf(m, v[a]); }
  float s = 0.0f;
#pragma unroll
  for (int a = 0; a < N_; ++a) { v[a] = __expf(v[a] - m); s += v[a]; }
  const float inv = 1.0f / s;
#pragma unroll
  for (int a = 0; a < N_; ++a) base[a * NN_] = v[a] * inv;
}

// ---------------------------------------------------------------------------
// Kernel 4: x[b,x,y,h,d] = sum_a att[b,h,a,x,y] * lv[b,x,a,h,d] * rv[b,a,y,h,d]
// Triple product (rv couples a,y,d) -> not a GEMM; FMA wave per (b,h,x,y),
// lane = d. Result stored f16 so the output projection consumes it directly.
// ---------------------------------------------------------------------------
__global__ void valsum_kernel(const float* __restrict__ lv32,
                              const float* __restrict__ rv32,
                              const float* __restrict__ att,
                              _Float16* __restrict__ xbuf16) {
  const int gid  = blockIdx.x * blockDim.x + threadIdx.x;
  const int wave = gid >> 5;
  const int lane = gid & 31;                       // d within head

  const int y = wave & (N_ - 1);
  const int x = (wave >> 6) & (N_ - 1);
  const int h = (wave >> 12) & (H_ - 1);
  const int b = wave >> 15;

  const float* attp = att + (size_t)((b * H_ + h) * N_) * NN_ + x * N_ + y;
  const float* lvp  = lv32 + (size_t)((b * N_ + x) * N_) * D_ + h * DK_ + lane;
  const float* rvp  = rv32 + ((size_t)(b * N_) * N_ + y) * D_ + h * DK_ + lane;

  float acc = 0.0f;
#pragma unroll 4
  for (int a = 0; a < N_; ++a) {
    const float w = attp[(size_t)a * NN_];          // wave-uniform scalar
    const float l = lvp[(size_t)a * D_];
    const float r = rvp[(size_t)a * (N_ * D_)];
    acc = fmaf(w * l, r, acc);
  }
  xbuf16[((size_t)(b * N_ + x) * N_ + y) * D_ + h * DK_ + lane] = (_Float16)acc;
}

// ---------------------------------------------------------------------------
// Kernel 5: out = xbuf @ W_o^T + b_o  (WMMA GEMM, f16 operands, f32 out)
// ---------------------------------------------------------------------------
__global__ void outproj_kernel(const _Float16* __restrict__ xbuf16,
                               const _Float16* __restrict__ Wo16,
                               const float* __restrict__ bo,
                               float* __restrict__ out) {
  const int wave = (blockIdx.x * blockDim.x + threadIdx.x) >> 5;
  const int lane = threadIdx.x & 31;
  const int half = lane >> 4;
  const int l16  = lane & 15;

  const int tile_n = wave & 15;
  const int tile_m = wave >> 4;

  const int arow = tile_m * 16 + l16;
  const int ecol = tile_n * 16 + l16;

  const _Float16* aRow = xbuf16 + (size_t)arow * D_;
  const _Float16* wRow = Wo16 + (size_t)ecol * D_;

  v8f c = {};
#pragma unroll
  for (int kk = 0; kk < D_ / 32; ++kk) {
    const int k0 = kk * 32;
    const v16h a  = load_a_frag(aRow, k0, half);
    const v16h bf = load_b_frag(wRow, k0, half);
    c = wmma_f16(a, bf, c);
  }

  const float bias = bo[ecol];
#pragma unroll
  for (int v = 0; v < 8; ++v) {
    const int orow = tile_m * 16 + half * 8 + v;
    out[(size_t)orow * D_ + ecol] = c[v] + bias;
  }
}

// ---------------------------------------------------------------------------
// Host launch
// ---------------------------------------------------------------------------
extern "C" void kernel_launch(void* const* d_in, const int* in_sizes, int n_in,
                              void* d_out, int out_size, void* d_ws, size_t ws_size,
                              hipStream_t stream) {
  const float*         state = (const float*)d_in[0];
  const unsigned char* mask  = (const unsigned char*)d_in[1];
  const float* W_lk = (const float*)d_in[2];
  const float* b_lk = (const float*)d_in[3];
  const float* W_rk = (const float*)d_in[4];
  const float* b_rk = (const float*)d_in[5];
  const float* W_lv = (const float*)d_in[6];
  const float* b_lv = (const float*)d_in[7];
  const float* W_rv = (const float*)d_in[8];
  const float* b_rv = (const float*)d_in[9];
  const float* W_o  = (const float*)d_in[10];
  const float* b_o  = (const float*)d_in[11];
  float* out = (float*)d_out;

  // Workspace layout. All section sizes are multiples of 16 bytes.
  const size_t PROJ = (size_t)ROWS_ * D_;            // 2,097,152 elements
  const size_t ATT  = (size_t)B_ * H_ * N_ * NN_;    // 4,194,304 floats
  _Float16* state16 = (_Float16*)d_ws;
  _Float16* Wlk16   = state16 + PROJ;
  _Float16* Wrk16   = Wlk16 + DD_;
  _Float16* Wlv16   = Wrk16 + DD_;
  _Float16* Wrv16   = Wlv16 + DD_;
  _Float16* Wo16    = Wrv16 + DD_;
  _Float16* lk16    = Wo16 + DD_;
  _Float16* rk16    = lk16 + PROJ;
  _Float16* xbuf16  = rk16 + PROJ;
  float*    lv32    = (float*)(xbuf16 + PROJ);
  float*    rv32    = lv32 + PROJ;
  float*    att     = rv32 + PROJ;

  const int TPB = 256;  // 8 wave32s per block

  // 0) Pack fp32 -> f16: state + 5 weight matrices
  pack_f16_kernel<<<(int)(PROJ / 4 + TPB - 1) / TPB, TPB, 0, stream>>>(
      state, state16, (int)(PROJ / 4));
  pack_f16_kernel<<<(DD_ / 4 + TPB - 1) / TPB, TPB, 0, stream>>>(W_lk, Wlk16, DD_ / 4);
  pack_f16_kernel<<<(DD_ / 4 + TPB - 1) / TPB, TPB, 0, stream>>>(W_rk, Wrk16, DD_ / 4);
  pack_f16_kernel<<<(DD_ / 4 + TPB - 1) / TPB, TPB, 0, stream>>>(W_lv, Wlv16, DD_ / 4);
  pack_f16_kernel<<<(DD_ / 4 + TPB - 1) / TPB, TPB, 0, stream>>>(W_rv, Wrv16, DD_ / 4);
  pack_f16_kernel<<<(DD_ / 4 + TPB - 1) / TPB, TPB, 0, stream>>>(W_o, Wo16, DD_ / 4);

  // 1) Four projections: 8192 tiles -> 8192 waves
  proj4_kernel<<<(ROWS_ / 16) * (D_ / 16) * 32 / TPB, TPB, 0, stream>>>(
      state16, Wlk16, Wrk16, Wlv16, Wrv16, b_lk, b_rk, b_lv, b_rv,
      lk16, rk16, lv32, rv32);

  // 2) Scores: B*H*N slabs * 16 tiles -> 16384 waves
  scores_kernel<<<(B_ * H_ * N_ * 16) * 32 / TPB, TPB, 0, stream>>>(
      lk16, rk16, mask, att);

  // 3) Softmax over a: 65536 threads
  softmax_kernel<<<(B_ * H_ * N_ * N_) / TPB, TPB, 0, stream>>>(att);

  // 4) Weighted triple-product sum: 65536 waves
  valsum_kernel<<<(B_ * H_ * N_ * N_) * 32 / TPB, TPB, 0, stream>>>(
      lv32, rv32, att, xbuf16);

  // 5) Output projection into d_out: 8192 waves
  outproj_kernel<<<(ROWS_ / 16) * (D_ / 16) * 32 / TPB, TPB, 0, stream>>>(
      xbuf16, Wo16, b_o, out);
}
